// GCL_29875792511391
// MI455X (gfx1250) — compile-verified, matrix operand order checked by reading, not run on hardware
//
#include <hip/hip_runtime.h>
#include <hip/hip_bf16.h>
#include <math.h>

// CDNA5 / gfx1250: wave32, WMMA 16x16x32 bf16 path.
typedef __attribute__((ext_vector_type(16))) __bf16 v16bf;
typedef __attribute__((ext_vector_type(8)))  float  v8f;

#define FDIM 128
#define WAVES 8
#define TILE_E 16

__device__ __forceinline__ float sigmoidf_(float x) {
    return 1.0f / (1.0f + __expf(-x));
}
__device__ __forceinline__ float softsignf_(float x) {
    return x / (1.0f + fabsf(x));
}

// ---------------------------------------------------------------------------
// Pre-pass: convert fp32 weight [Kdim][128] into bf16 laid out exactly as the
// WMMA B operand wants it: out[((kc*8 + t)*32 + lane)*16 + j], where
//   K = (j<8 ? j : 16+(j-8)) + (lane>=16 ? 8 : 0), gk = kc*32 + K,
//   n = t*16 + (lane&15)
// (16-bit B 32x16 VGPR layout per CDNA5 ISA 7.12.2). Each lane then fetches
// its whole operand as one contiguous 32-byte load.
// ---------------------------------------------------------------------------
__global__ void swizzle_w(const float* __restrict__ W, __bf16* __restrict__ out, int Kdim) {
    int idx = blockIdx.x * blockDim.x + threadIdx.x;
    int total = Kdim * FDIM;
    if (idx >= total) return;
    int j    = idx & 15;
    int lane = (idx >> 4) & 31;
    int t    = (idx >> 9) & 7;
    int kc   = idx >> 12;
    int K  = ((j < 8) ? j : 16 + (j - 8)) + ((lane >= 16) ? 8 : 0);
    int gk = kc * 32 + K;
    int nn = t * 16 + (lane & 15);
    out[idx] = (__bf16)W[gk * FDIM + nn];
}

__global__ void zero_f32(float* __restrict__ p, int n) {
    int i = blockIdx.x * blockDim.x + threadIdx.x;
    if (i < n) p[i] = 0.0f;
}

// ---------------------------------------------------------------------------
// Edge kernel: 8 waves/block, 16 edges per wave.
//   h   = sigmoid([src|dst] @ Wm1 + bm1)      (16x256 @ 256x128, 64 WMMAs)
//   msg = softsign(h @ Wm2 + bm2)             (16x128 @ 128x128, 32 WMMAs)
//   agg[rows[e]] += msg                       (f32 atomics, L2-resident)
// ---------------------------------------------------------------------------
__global__ __launch_bounds__(256) void edge_msg(
    const float* __restrict__ features,
    const int*   __restrict__ rows,
    const int*   __restrict__ cols,
    const __bf16* __restrict__ Wm1s,   // [8][8][32][16] bf16
    const float* __restrict__ bm1,
    const __bf16* __restrict__ Wm2s,   // [4][8][32][16] bf16
    const float* __restrict__ bm2,
    float* __restrict__ agg,
    int n_edges)
{
    __shared__ __bf16 h_lds[WAVES][TILE_E][FDIM];   // 32 KB, wave-private tiles

    const int lane = threadIdx.x & 31;
    const int wave = threadIdx.x >> 5;
    const int e_base = blockIdx.x * (WAVES * TILE_E) + wave * TILE_E;
    if (e_base >= n_edges) return;                  // per-wave guard, no block sync used

    const int m    = lane & 15;                      // edge-in-tile for A operand
    const int koff = (lane >> 4) * 8;                // A/B lane K offset

    const int rsrc = rows[e_base + m];
    const int csrc = cols[e_base + m];
    const float* fsrc = features + (long)rsrc * FDIM;
    const float* fdst = features + (long)csrc * FDIM;

    // Preload A: 16 edges x 256 (concat src|dst), bf16, 8 K-chunks of 32.
    v16bf A[8];
    #pragma unroll
    for (int kc = 0; kc < 8; ++kc) {
        const float* src = (kc < 4) ? fsrc : fdst;
        const int cb = kc * 32 - ((kc < 4) ? 0 : 128) + koff;
        #pragma unroll
        for (int v = 0; v < 8; ++v) {
            int kb = cb + ((v >= 4) ? 16 + 2 * (v - 4) : 2 * v);   // consecutive pair
            float2 f = *(const float2*)(src + kb);
            A[kc][2 * v]     = (__bf16)f.x;
            A[kc][2 * v + 1] = (__bf16)f.y;
        }
    }

    // Layer 1: h = sigmoid(A @ Wm1 + bm1) -> stage to LDS in [m][n] order.
    const int mrow = (lane >> 4) * 8;
    #pragma unroll
    for (int t = 0; t < 8; ++t) {
        v8f acc = {};
        #pragma unroll
        for (int kc = 0; kc < 8; ++kc) {
            v16bf b = *(const v16bf*)(Wm1s + (((kc * 8 + t) * 32 + lane) << 4));
            acc = __builtin_amdgcn_wmma_f32_16x16x32_bf16(
                    false, A[kc], false, b, (short)0, acc, false, false);
        }
        int n = t * 16 + (lane & 15);
        float bias = bm1[n];
        #pragma unroll
        for (int r = 0; r < 8; ++r)
            h_lds[wave][mrow + r][n] = (__bf16)sigmoidf_(acc[r] + bias);
    }
    // Same-wave LDS is in-order; barrier the counter + compiler reorder.
    asm volatile("s_wait_dscnt 0x0" ::: "memory");

    // Reload h as A operand (D layout -> A layout transpose via LDS).
    v16bf A2[4];
    #pragma unroll
    for (int kc = 0; kc < 4; ++kc)
        #pragma unroll
        for (int j = 0; j < 16; ++j) {
            int k = kc * 32 + ((j < 8) ? j : 16 + (j - 8)) + koff;
            A2[kc][j] = h_lds[wave][m][k];
        }

    // Edge row ids for the rows this lane owns in the D layout.
    int rn[8];
    #pragma unroll
    for (int r = 0; r < 8; ++r) rn[r] = rows[e_base + mrow + r];

    // Layer 2 + scatter-add.
    #pragma unroll
    for (int t = 0; t < 8; ++t) {
        v8f acc = {};
        #pragma unroll
        for (int kc = 0; kc < 4; ++kc) {
            v16bf b = *(const v16bf*)(Wm2s + (((kc * 8 + t) * 32 + lane) << 4));
            acc = __builtin_amdgcn_wmma_f32_16x16x32_bf16(
                    false, A2[kc], false, b, (short)0, acc, false, false);
        }
        int n = t * 16 + (lane & 15);
        float bias = bm2[n];
        #pragma unroll
        for (int r = 0; r < 8; ++r) {
            float vmsg = softsignf_(acc[r] + bias);
            atomicAdd(agg + (long)rn[r] * FDIM + n, vmsg);
        }
    }
}

// ---------------------------------------------------------------------------
// Node kernel: g = sigmoid([feat|agg|te]); h = sigmoid(g@Wf1+bf1);
//              out = softsign(h@Wf2+bf2). 16 nodes per wave.
// ---------------------------------------------------------------------------
__global__ __launch_bounds__(256) void node_mlp(
    const float* __restrict__ features,
    const float* __restrict__ agg,
    const float* __restrict__ te,
    const __bf16* __restrict__ Wf1s,   // [12][8][32][16] bf16
    const float* __restrict__ bf1,
    const __bf16* __restrict__ Wf2s,   // [4][8][32][16] bf16
    const float* __restrict__ bf2,
    float* __restrict__ out,
    int n_tiles)
{
    __shared__ __bf16 h_lds[WAVES][TILE_E][FDIM];

    const int lane = threadIdx.x & 31;
    const int wave = threadIdx.x >> 5;
    int tile = blockIdx.x * WAVES + wave;
    const bool active = tile < n_tiles;
    if (!active) tile = 0;                          // compute redundantly, skip stores
    const int node_base = tile * TILE_E;

    const int m    = lane & 15;
    const int koff = (lane >> 4) * 8;
    const float* f0 = features + (long)(node_base + m) * FDIM;
    const float* f1 = agg      + (long)(node_base + m) * FDIM;
    const float* f2 = te       + (long)(node_base + m) * FDIM;

    // A = sigmoid(concat) as bf16, 12 K-chunks of 32 (K = 384).
    v16bf A[12];
    #pragma unroll
    for (int kc = 0; kc < 12; ++kc) {
        const float* src = (kc < 4) ? f0 : ((kc < 8) ? f1 : f2);
        const int cb = (kc & 3) * 32 + koff;
        #pragma unroll
        for (int v = 0; v < 8; ++v) {
            int kb = cb + ((v >= 4) ? 16 + 2 * (v - 4) : 2 * v);
            float2 f = *(const float2*)(src + kb);
            A[kc][2 * v]     = (__bf16)sigmoidf_(f.x);
            A[kc][2 * v + 1] = (__bf16)sigmoidf_(f.y);
        }
    }

    const int mrow = (lane >> 4) * 8;
    #pragma unroll
    for (int t = 0; t < 8; ++t) {
        v8f acc = {};
        #pragma unroll
        for (int kc = 0; kc < 12; ++kc) {
            v16bf b = *(const v16bf*)(Wf1s + (((kc * 8 + t) * 32 + lane) << 4));
            acc = __builtin_amdgcn_wmma_f32_16x16x32_bf16(
                    false, A[kc], false, b, (short)0, acc, false, false);
        }
        int n = t * 16 + (lane & 15);
        float bias = bf1[n];
        #pragma unroll
        for (int r = 0; r < 8; ++r)
            h_lds[wave][mrow + r][n] = (__bf16)sigmoidf_(acc[r] + bias);
    }
    asm volatile("s_wait_dscnt 0x0" ::: "memory");

    v16bf A2[4];
    #pragma unroll
    for (int kc = 0; kc < 4; ++kc)
        #pragma unroll
        for (int j = 0; j < 16; ++j) {
            int k = kc * 32 + ((j < 8) ? j : 16 + (j - 8)) + koff;
            A2[kc][j] = h_lds[wave][m][k];
        }

    #pragma unroll
    for (int t = 0; t < 8; ++t) {
        v8f acc = {};
        #pragma unroll
        for (int kc = 0; kc < 4; ++kc) {
            v16bf b = *(const v16bf*)(Wf2s + (((kc * 8 + t) * 32 + lane) << 4));
            acc = __builtin_amdgcn_wmma_f32_16x16x32_bf16(
                    false, A2[kc], false, b, (short)0, acc, false, false);
        }
        int n = t * 16 + (lane & 15);
        float bias = bf2[n];
        if (active) {
            #pragma unroll
            for (int r = 0; r < 8; ++r)
                out[(long)(node_base + mrow + r) * FDIM + n] = softsignf_(acc[r] + bias);
        }
    }
}

// ---------------------------------------------------------------------------
extern "C" void kernel_launch(void* const* d_in, const int* in_sizes, int n_in,
                              void* d_out, int out_size, void* d_ws, size_t ws_size,
                              hipStream_t stream) {
    const float* features = (const float*)d_in[0];
    const int*   rows     = (const int*)d_in[1];
    const int*   cols     = (const int*)d_in[2];
    const float* te       = (const float*)d_in[3];
    const float* Wm1      = (const float*)d_in[4];
    const float* bm1      = (const float*)d_in[5];
    const float* Wm2      = (const float*)d_in[6];
    const float* bm2      = (const float*)d_in[7];
    const float* Wf1      = (const float*)d_in[8];
    const float* bf1      = (const float*)d_in[9];
    const float* Wf2      = (const float*)d_in[10];
    const float* bf2      = (const float*)d_in[11];
    float* out = (float*)d_out;

    const int n_nodes = in_sizes[0] / FDIM;   // 10000
    const int n_edges = in_sizes[1];          // 640000

    // Workspace layout: bf16 swizzled weights, then 256B-aligned f32 agg.
    __bf16* wm1s = (__bf16*)d_ws;             // 256*128
    __bf16* wm2s = wm1s + 256 * FDIM;         // 128*128
    __bf16* wf1s = wm2s + 128 * FDIM;         // 384*128
    __bf16* wf2s = wf1s + 384 * FDIM;         // 128*128
    size_t wbytes = (size_t)(256 + 128 + 384 + 128) * FDIM * sizeof(__bf16); // 229376 (256B-mult)
    float* agg = (float*)((char*)d_ws + wbytes);

    swizzle_w<<<(256 * FDIM + 255) / 256, 256, 0, stream>>>(Wm1, wm1s, 256);
    swizzle_w<<<(128 * FDIM + 255) / 256, 256, 0, stream>>>(Wm2, wm2s, 128);
    swizzle_w<<<(384 * FDIM + 255) / 256, 256, 0, stream>>>(Wf1, wf1s, 384);
    swizzle_w<<<(128 * FDIM + 255) / 256, 256, 0, stream>>>(Wf2, wf2s, 128);
    zero_f32<<<(n_nodes * FDIM + 255) / 256, 256, 0, stream>>>(agg, n_nodes * FDIM);

    int eblocks = (n_edges + WAVES * TILE_E - 1) / (WAVES * TILE_E);   // 5000
    edge_msg<<<eblocks, 256, 0, stream>>>(features, rows, cols,
                                          wm1s, bm1, wm2s, bm2, agg, n_edges);

    int n_tiles = (n_nodes + TILE_E - 1) / TILE_E;                      // 625
    node_mlp<<<(n_tiles + WAVES - 1) / WAVES, 256, 0, stream>>>(
        features, agg, te, wf1s, bf1, wf2s, bf2, out, n_tiles);
}
